// RGCN_90640989815340
// MI455X (gfx1250) — matrix operand order repaired and checked by reading, not compiled, not run on hardware
//
#include <hip/hip_runtime.h>
#include <hip/hip_bf16.h>
#include <math.h>

// ---------------- problem constants ----------------
#define N_NODES   100000
#define N_EDGES   500000
#define NUM_REL   8
#define HID       128
#define KDIM      1152            // R*HID + HID (root folded into GEMM)
#define KT        36              // KDIM / 32  (bf16 WMMA k-step = 32)
#define MT        6250            // N_NODES / 16 (exact)
#define B_ELEMS   (KDIM * HID)    // 147456 bf16 = 288 KB
#define LDS_BYTES (B_ELEMS * 2)

typedef __attribute__((ext_vector_type(16))) __bf16 bf16x16;
typedef __attribute__((ext_vector_type(8)))  float  v8f;

static __device__ __forceinline__ unsigned short f2bf(float f) {
    union { float f; unsigned u; } v; v.f = f;
    unsigned u = v.u;
    unsigned r = (u + 0x7fffu + ((u >> 16) & 1u)) >> 16;   // RNE
    return (unsigned short)r;
}

// ---------------- stage 0: pack B = [W (8x128x128) ; root (128x128)] into
// WMMA B-fragment order: [ntile(8)][ktile(36)][lane(32)][elem(16)] bf16 -----
__global__ void rgcn_pack_B(const float* __restrict__ W,
                            const float* __restrict__ root,
                            unsigned short* __restrict__ Bp) {
    int p = blockIdx.x * 256 + threadIdx.x;          // < 8*36*32*16 = 147456
    int e    = p & 15;
    int lane = (p >> 4) & 31;
    int kt   = (p >> 9) % KT;
    int nt   = p / (KT * 512);
    int col  = nt * 16 + (lane & 15);
    int k    = kt * 32 + (e & 7) + (lane >> 4) * 8 + (e >> 3) * 16;       // B: K table order {0..7,16..23}/{8..15,24..31} per lane-half
    float val = (k < NUM_REL * HID)
        ? W[(size_t)k * HID + col]                   // W[r][d][f] flattened: k=r*128+d
        : root[(size_t)(k - NUM_REL * HID) * HID + col];
    Bp[p] = f2bf(val);
}

// ---------------- stage 1: embedding gather x = emb[x_ids] ----------------
__global__ void rgcn_gather(const int* __restrict__ ids,
                            const float* __restrict__ emb,
                            float* __restrict__ x) {
    int t = blockIdx.x * 256 + threadIdx.x;          // < N*128
    x[t] = emb[(size_t)ids[t >> 7] * HID + (t & 127)];
}

// ---------------- stage 2: per-(dst,rel) edge counts ----------------------
__global__ void rgcn_count(const int* __restrict__ ei,
                           const int* __restrict__ et,
                           int* __restrict__ counts) {
    int e = blockIdx.x * 256 + threadIdx.x;
    if (e >= N_EDGES) return;
    int dst = ei[N_EDGES + e];
    int r   = et[e];
    atomicAdd(&counts[dst * NUM_REL + r], 1);
}

// ---------------- stage 3: scatter-add raw features into [N, R*128] f32 ---
__global__ void rgcn_scatter(const int* __restrict__ ei,
                             const int* __restrict__ et,
                             const float* __restrict__ x,
                             float* __restrict__ sums) {
    int gid = blockIdx.x * 256 + threadIdx.x;        // < E*32
    int e  = gid >> 5;
    int ln = gid & 31;                               // 32 lanes * 4 floats = 128 dims
    int src = ei[e];
    int dst = ei[N_EDGES + e];
    int r   = et[e];
    const float* xs = x + (size_t)src * HID + ln * 4;
    float* sp = sums + (size_t)dst * (NUM_REL * HID) + r * HID + ln * 4;
    atomicAdd(sp + 0, xs[0]);
    atomicAdd(sp + 1, xs[1]);
    atomicAdd(sp + 2, xs[2]);
    atomicAdd(sp + 3, xs[3]);
}

// ---------------- stage 4: normalize + pack A-fragments -------------------
// A (bf16, [N,1152]) stored in WMMA A-frag order: [mtile][ktile][lane][elem]
// 16-bit A 16x32 layout: lane<16 holds K in {0..7,16..23}, lane>=16 {8..15,24..31}
__global__ void rgcn_pack_A(const float* __restrict__ sums,
                            const int* __restrict__ counts,
                            const float* __restrict__ x,
                            unsigned short* __restrict__ Ap) {
    size_t p = (size_t)blockIdx.x * 256 + threadIdx.x;   // < N * 1152
    int e    = (int)(p & 15);
    int lane = (int)((p >> 4) & 31);
    int kt   = (int)((p >> 9) % KT);
    size_t mt = p / (KT * 512);
    int half = lane >> 4;
    int kk = (e & 7) + half * 8 + ((e >> 3) * 16);
    int k  = kt * 32 + kk;
    size_t m = mt * 16 + (lane & 15);
    float val;
    if (k < NUM_REL * HID) {
        int r   = k >> 7;
        int cnt = counts[m * NUM_REL + r];
        val = sums[m * (NUM_REL * HID) + k] * (1.0f / (float)(cnt > 0 ? cnt : 1));
    } else {
        val = x[m * HID + (k - NUM_REL * HID)];
    }
    Ap[p] = f2bf(val);
}

// ---------------- stage 5: GEMM [N,1152]x[1152,128] via bf16 WMMA ---------
// B staged in LDS (288 KB, shared by 8 waves). K-loop fully unrolled with a
// 2-deep ping-pong register buffer: rotation becomes SSA renaming (no moves,
// no WMMA->VALU hazard NOPs). One wave = 16x128 strip, 8 accumulators.
__global__ void __launch_bounds__(256, 1)
rgcn_gemm(const unsigned short* __restrict__ Apack,
          const unsigned short* __restrict__ Bpack,
          const float* __restrict__ bias,
          float* __restrict__ out,
          int act) {                                  // 0 = relu, 1 = sigmoid
    extern __shared__ unsigned short Bs[];            // B_ELEMS ushorts = 288 KB

    // cooperative stage of entire packed B into LDS (72 x b128 per thread)
    {
        const uint4* src = (const uint4*)Bpack;
        uint4*       dst = (uint4*)Bs;
        #pragma unroll 4
        for (int i = threadIdx.x; i < B_ELEMS / 8; i += 256)
            dst[i] = src[i];
    }
    __syncthreads();

    int wave = threadIdx.x >> 5;
    int lane = threadIdx.x & 31;
    int mtile = blockIdx.x * 8 + wave;
    if (mtile >= MT) return;                          // wave-uniform: EXEC stays all-1

    const bf16x16* A  = (const bf16x16*)Apack;
    const bf16x16* Bl = (const bf16x16*)Bs;

    v8f acc[8];
    #pragma unroll
    for (int j = 0; j < 8; ++j) acc[j] = (v8f){};

    size_t abase = (size_t)mtile * KT * 32 + lane;

    // 2-deep ping-pong operand buffers; full unroll -> pure SSA renaming
    bf16x16 a_buf[2];
    bf16x16 b_buf[2][8];

    a_buf[0] = A[abase];
    #pragma unroll
    for (int j = 0; j < 8; ++j)
        b_buf[0][j] = Bl[(size_t)j * KT * 32 + lane];

    #pragma unroll
    for (int kt = 0; kt < KT; ++kt) {
        const int cur = kt & 1;
        const int nxt = cur ^ 1;
        if (kt + 1 < KT) {
            a_buf[nxt] = A[abase + (size_t)(kt + 1) * 32];
            #pragma unroll
            for (int j = 0; j < 8; ++j)
                b_buf[nxt][j] = Bl[((size_t)j * KT + kt + 1) * 32 + lane];
        }
        #pragma unroll
        for (int j = 0; j < 8; ++j)
            acc[j] = __builtin_amdgcn_wmma_f32_16x16x32_bf16(
                false, a_buf[cur], false, b_buf[cur][j], (short)0, acc[j],
                false, false);
    }

    // D layout (f32 16x16): lanes 0-15 -> M = v, lanes 16-31 -> M = v+8; N = lane%16
    int half = lane >> 4;
    int coll = lane & 15;
    #pragma unroll
    for (int j = 0; j < 8; ++j) {
        int col = j * 16 + coll;
        float bcol = bias[col];
        #pragma unroll
        for (int v = 0; v < 8; ++v) {
            int row = mtile * 16 + v + half * 8;
            float val = acc[j][v] + bcol;
            if (act == 0) {
                val = val > 0.0f ? val : 0.0f;
            } else {
                val = 1.0f / (1.0f + __expf(-val));
            }
            out[(size_t)row * HID + col] = val;
        }
    }
}

// ---------------- host-side layer driver ----------------------------------
static void run_layer(const float* x, const unsigned short* Bp, const float* bias,
                      float* dst, int act,
                      const int* ei, const int* et,
                      float* sums, int* counts, unsigned short* Apack,
                      hipStream_t stream) {
    hipMemsetAsync(sums,   0, (size_t)N_NODES * NUM_REL * HID * sizeof(float), stream);
    hipMemsetAsync(counts, 0, (size_t)N_NODES * NUM_REL * sizeof(int), stream);
    rgcn_count  <<<(N_EDGES + 255) / 256, 256, 0, stream>>>(ei, et, counts);
    rgcn_scatter<<<(N_EDGES * 32) / 256, 256, 0, stream>>>(ei, et, x, sums);
    rgcn_pack_A <<<((size_t)N_NODES * KDIM) / 256, 256, 0, stream>>>(sums, counts, x, Apack);
    rgcn_gemm   <<<(MT + 7) / 8, 256, LDS_BYTES, stream>>>(Apack, Bp, bias, dst, act);
}

extern "C" void kernel_launch(void* const* d_in, const int* in_sizes, int n_in,
                              void* d_out, int out_size, void* d_ws, size_t ws_size,
                              hipStream_t stream) {
    const int*   x_ids = (const int*)  d_in[0];
    const int*   ei    = (const int*)  d_in[1];   // [2, E]
    const int*   et    = (const int*)  d_in[2];   // [E]
    const float* emb   = (const float*)d_in[3];
    const float* W1    = (const float*)d_in[4];
    const float* root1 = (const float*)d_in[5];
    const float* b1    = (const float*)d_in[6];
    const float* W2    = (const float*)d_in[7];
    const float* root2 = (const float*)d_in[8];
    const float* b2    = (const float*)d_in[9];
    float* out = (float*)d_out;

    char* ws = (char*)d_ws;
    size_t off = 0;
    auto carve = [&](size_t bytes) {
        void* p = ws + off;
        off = (off + bytes + 255) & ~(size_t)255;
        return p;
    };
    float*          sums   = (float*)carve((size_t)N_NODES * NUM_REL * HID * sizeof(float)); // 409.6 MB
    int*            counts = (int*)  carve((size_t)N_NODES * NUM_REL * sizeof(int));         //   3.2 MB
    float*          x1     = (float*)carve((size_t)N_NODES * HID * sizeof(float));           //  51.2 MB
    float*          x2     = (float*)carve((size_t)N_NODES * HID * sizeof(float));           //  51.2 MB
    unsigned short* Apack  = (unsigned short*)carve((size_t)N_NODES * KDIM * 2);             // 230.4 MB
    unsigned short* B1p    = (unsigned short*)carve((size_t)KDIM * HID * 2);                 // 294 KB
    unsigned short* B2p    = (unsigned short*)carve((size_t)KDIM * HID * 2);                 // 294 KB
    (void)ws_size; (void)n_in; (void)in_sizes; (void)out_size;

    // allow 288 KB dynamic LDS for the GEMM (WGP has 320 KB)
    hipFuncSetAttribute((const void*)rgcn_gemm,
                        hipFuncAttributeMaxDynamicSharedMemorySize, LDS_BYTES);

    // weight packing (tiny, every call -> deterministic)
    rgcn_pack_B<<<(KDIM * HID) / 256, 256, 0, stream>>>(W1, root1, B1p);
    rgcn_pack_B<<<(KDIM * HID) / 256, 256, 0, stream>>>(W2, root2, B2p);

    // x = emb[x_ids]
    rgcn_gather<<<((size_t)N_NODES * HID) / 256, 256, 0, stream>>>(x_ids, emb, x1);

    // layer 1: relu(RGCNConv(x1)) -> x2
    run_layer(x1, B1p, b1, x2, /*relu*/0, ei, et, sums, counts, Apack, stream);
    // layer 2: sigmoid(RGCNConv(x2)) -> out
    run_layer(x2, B2p, b2, out, /*sigmoid*/1, ei, et, sums, counts, Apack, stream);
}
